// MotifAwareAttention_52536039964868
// MI455X (gfx1250) — matrix operand order repaired and checked by reading, not compiled, not run on hardware
//
#include <hip/hip_runtime.h>
#include <hip/hip_bf16.h>
#include <math.h>

// Problem constants (match reference)
#define B_    2
#define T_    2048
#define C_    1024
#define H_    16
#define HD_   64
#define M_    (B_ * T_)   // 4096 rows
#define NQKV  (3 * C_)    // 3072

typedef __attribute__((ext_vector_type(16))) __bf16 v16bf;
typedef __attribute__((ext_vector_type(8)))  __bf16 v8bf;
typedef __attribute__((ext_vector_type(8)))  float  v8f;
typedef __attribute__((ext_vector_type(4)))  unsigned int u32x4;
typedef __attribute__((ext_vector_type(4)))  int          i32x4;
typedef __attribute__((ext_vector_type(8)))  int          i32x8;

union V16U { v16bf v; v8bf h[2]; };

__device__ __forceinline__ unsigned short f2bf(float f) {
  unsigned int u = __float_as_uint(f);
  unsigned int r = u + 0x7FFFu + ((u >> 16) & 1u);   // round-to-nearest-even
  return (unsigned short)(r >> 16);
}

__device__ __forceinline__ v8f wmma_bf16(v16bf a, v16bf b, v8f c) {
  // v_wmma_f32_16x16x32_bf16  (neg_a, A, neg_b, B, c_mod, C, reuse_a, reuse_b)
  return __builtin_amdgcn_wmma_f32_16x16x32_bf16(false, a, false, b, (short)0, c,
                                                 false, false);
}

// ------------------------------------------------------------------ TDM
// Issue a 2-D tensor_load_to_lds: tile_d1 rows of tile_d0 contiguous bf16
// elements, row pitch stride_d0 elements.  D# layout per CDNA5 ISA 8.3/8.4:
// group0 = {count=1 | lds_addr | global_addr | type=2}, group1 carries
// data_size=1 (2 bytes), tensor dims (set generously; tiles are in-bounds),
// tile dims and dim0 stride.  Tracked by TENSORcnt.
__device__ __forceinline__ void tdm_load_2d(void* lds_dst, const void* gsrc,
                                            unsigned tile_d0, unsigned tile_d1,
                                            unsigned stride_d0) {
  unsigned long long ga = (unsigned long long)(uintptr_t)gsrc;
  unsigned lds = (unsigned)(uintptr_t)lds_dst;   // low 32 bits = LDS offset
  u32x4 g0;
  g0[0] = 1u;                                          // count=1, user mode
  g0[1] = lds;                                         // lds_addr (bytes)
  g0[2] = (unsigned)(ga & 0xFFFFFFFFu);                // global_addr[31:0]
  g0[3] = (unsigned)((ga >> 32) & 0x01FFFFFFu) | 0x80000000u;  // addr hi | type=2
  const unsigned td0 = 1u << 20, td1 = 1u << 20;       // tensor dims (OOB off)
  i32x8 g1;
  g1[0] = (int)(1u << 16);                             // data_size=1 (2B), mask=0
  g1[1] = (int)((td0 & 0xFFFFu) << 16);                // tensor_dim0[15:0]
  g1[2] = (int)(((td0 >> 16) & 0xFFFFu) | ((td1 & 0xFFFFu) << 16));
  g1[3] = (int)(((td1 >> 16) & 0xFFFFu) | ((tile_d0 & 0xFFFFu) << 16));
  g1[4] = (int)(tile_d1 & 0xFFFFu);                    // tile_dim1, tile_dim2=0
  g1[5] = (int)stride_d0;                              // tensor_dim0_stride[31:0]
  g1[6] = 0;
  g1[7] = 0;
  i32x4 z4 = {0, 0, 0, 0};
#if defined(__clang_major__) && (__clang_major__ >= 23)
  i32x8 z8 = {0, 0, 0, 0, 0, 0, 0, 0};
  __builtin_amdgcn_tensor_load_to_lds(g0, g1, z4, z4, z8, 0);
#else
  __builtin_amdgcn_tensor_load_to_lds(g0, g1, z4, z4, 0);
#endif
}

// A fragment (16 x 32, bf16) from a row-major matrix; lane lr = row, hi selects
// K sub-chunks per ISA layout: halves0-7 = K hi*8..+7, halves8-15 = K 16+hi*8..+7
__device__ __forceinline__ v16bf load_a_frag(const unsigned short* base, int ld,
                                             int lr, int hi) {
  V16U u;
  const unsigned short* p = base + (size_t)lr * ld + hi * 8;
  u.h[0] = *(const v8bf*)(p);
  u.h[1] = *(const v8bf*)(p + 16);
  return u.v;
}

// B fragment (32 x 16, bf16) where element(k, n) = Wt[n][k] and Wt is row-major
// with row stride ld (per-column K contiguous). lane lr = column n,
// halves0-7 = K hi*8..+7, halves8-15 = K 16+hi*8..+7.
__device__ __forceinline__ v16bf load_b_frag(const unsigned short* base, int ld,
                                             int lr, int hi) {
  V16U u;
  const unsigned short* p = base + (size_t)lr * ld + hi * 8;
  u.h[0] = *(const v8bf*)(p);
  u.h[1] = *(const v8bf*)(p + 16);
  return u.v;
}

// ---------------------------------------------------------------- converts
__global__ void k_f32_to_bf16(const float* __restrict__ in,
                              unsigned short* __restrict__ out, int n) {
  int i = blockIdx.x * blockDim.x + threadIdx.x;
  int stride = gridDim.x * blockDim.x;
  for (; i < n; i += stride) out[i] = f2bf(in[i]);
}

// ---------------------------------------------------------------- GEMM
// Out[m,n] = sum_k A[m,k] * W[n,k]  (+ bias[n]).
// W tile (128 x 32) is staged into double-buffered LDS by the Tensor Data
// Mover (wave 0 issues, TENSORcnt + barrier synchronize); B-fragments then
// come from LDS, A-fragments stream from global per wave.
// MODE 0: QKV epilogue -> scatter bf16 Q (scaled 1/8), K, V-transposed.
// MODE 1: plain fp32 store (output projection).
template <int MODE>
__global__ void k_gemm(const unsigned short* __restrict__ A,
                       const unsigned short* __restrict__ W,
                       const float* __restrict__ bias,
                       unsigned short* __restrict__ Qb,
                       unsigned short* __restrict__ Kb,
                       unsigned short* __restrict__ Vt,
                       float* __restrict__ Out,
                       int Kdim, int Ndim) {
  __shared__ unsigned short Wlds[2][128][32];

  const int lane = threadIdx.x & 31;
  const int wave = threadIdx.x >> 5;
  const int lr = lane & 15;
  const int hi = lane >> 4;
  const int mbase = blockIdx.y * 128 + wave * 16;
  const int nbase = blockIdx.x * 128;

  v8f acc[8];
#pragma unroll
  for (int j = 0; j < 8; ++j)
#pragma unroll
    for (int e = 0; e < 8; ++e) acc[j][e] = 0.0f;

  const unsigned short* Arow = A + (size_t)mbase * Kdim;
  const unsigned short* Wtile = W + (size_t)nbase * Kdim;

  if (wave == 0)
    tdm_load_2d(&Wlds[0][0][0], Wtile, 32, 128, Kdim);

  const int nk = Kdim / 32;
  for (int i = 0; i < nk; ++i) {
    const int k0 = i * 32;
    const int buf = i & 1;
    __builtin_amdgcn_s_wait_tensorcnt(0);   // no-op for waves with cnt==0
    __syncthreads();                        // publish LDS tile to all waves
    if (wave == 0 && i + 1 < nk)            // overlap next DMA with compute
      tdm_load_2d(&Wlds[buf ^ 1][0][0], Wtile + k0 + 32, 32, 128, Kdim);
    if (i + 1 < nk)
      __builtin_prefetch(Arow + k0 + 32, 0, 1);  // global_prefetch_b8

    v16bf af = load_a_frag(Arow + k0, Kdim, lr, hi);
#pragma unroll
    for (int j = 0; j < 8; ++j) {
      v16bf bf = load_b_frag(&Wlds[buf][j * 16][0], 32, lr, hi);  // ds_load_b128
      acc[j] = wmma_bf16(af, bf, acc[j]);
    }
    // WMMA consumption forces dscnt waits before the next top barrier, so the
    // buffer is fully read before it can be overwritten by the DMA issued in the next iteration.
  }

#pragma unroll
  for (int j = 0; j < 8; ++j) {
    const int o = nbase + j * 16 + lr;      // output column
    const float bv = bias[o];
    if (MODE == 0) {
      const int s = o >> 10;                 // 0=Q 1=K 2=V
      const int rem = o & 1023;
      const int h = rem >> 6;
      const int d = rem & 63;
#pragma unroll
      for (int v = 0; v < 8; ++v) {
        const int m = mbase + hi * 8 + v;    // C/D layout: row = hi*8+v
        const int b = m >> 11;               // / T_
        const int t = m & (T_ - 1);
        const int bh = b * H_ + h;
        const float val = acc[j][v] + bv;
        if (s == 0)
          Qb[((size_t)bh * T_ + t) * HD_ + d] = f2bf(val * 0.125f);  // 1/sqrt(64)
        else if (s == 1)
          Kb[((size_t)bh * T_ + t) * HD_ + d] = f2bf(val);
        else
          Vt[((size_t)bh * HD_ + d) * T_ + t] = f2bf(val);           // transposed
      }
    } else {
#pragma unroll
      for (int v = 0; v < 8; ++v) {
        const int m = mbase + hi * 8 + v;
        Out[(size_t)m * Ndim + o] = acc[j][v] + bv;
      }
    }
  }
}

// ---------------------------------------------------------------- attention
// Flash attention, causal. One wave owns 16 query rows; 8 waves = 128 queries
// per block. Key tiles of 32, K/V tiles staged into double-buffered LDS by
// the TDM. Motif bias is skipped: softmax is shift-invariant along the key
// axis, so it cancels exactly.
__global__ void k_attention(const unsigned short* __restrict__ Qb,
                            const unsigned short* __restrict__ Kb,
                            const unsigned short* __restrict__ Vt,
                            unsigned short* __restrict__ Obf) {
  __shared__ unsigned short Plds[8][16][32];  // per-wave P staging (bf16)
  __shared__ unsigned short Klds[2][32][64];  // keys x d
  __shared__ unsigned short Vlds[2][64][32];  // d x keys (V pre-transposed)

  const int lane = threadIdx.x & 31;
  const int wave = threadIdx.x >> 5;
  const int lr = lane & 15;
  const int hi = lane >> 4;

  const int nqblk = T_ / 128;                 // 16
  const int bh = blockIdx.x / nqblk;
  const int qblk = blockIdx.x % nqblk;
  const int qw = qblk * 128 + wave * 16;      // this wave's first query row

  const unsigned short* Qh = Qb + (size_t)bh * T_ * HD_;
  const unsigned short* Kh = Kb + (size_t)bh * T_ * HD_;
  const unsigned short* Vh = Vt + (size_t)bh * HD_ * T_;

  // Q fragments stay resident (16 x 64 = two K-chunks)
  v16bf qf0 = load_a_frag(Qh + (size_t)qw * HD_ + 0, HD_, lr, hi);
  v16bf qf1 = load_a_frag(Qh + (size_t)qw * HD_ + 32, HD_, lr, hi);

  v8f oacc[4];
#pragma unroll
  for (int j = 0; j < 4; ++j)
#pragma unroll
    for (int e = 0; e < 8; ++e) oacc[j][e] = 0.0f;

  float run_m[8], run_s[8];
#pragma unroll
  for (int v = 0; v < 8; ++v) { run_m[v] = -1e30f; run_s[v] = 0.0f; }

  const int kend = (qblk + 1) * 128;          // uniform across block
  const int nt = kend / 32;

  if (wave == 0) {
    tdm_load_2d(&Klds[0][0][0], Kh, 64, 32, HD_);       // 32 key rows x 64
    tdm_load_2d(&Vlds[0][0][0], Vh, 32, 64, T_);        // 64 d rows x 32 keys
  }

  for (int i = 0; i < nt; ++i) {
    const int k0 = i * 32;
    const int buf = i & 1;
    __builtin_amdgcn_s_wait_tensorcnt(0);
    __syncthreads();                          // K/V tile visible to all waves
    if (wave == 0 && i + 1 < nt) {            // overlap next DMA with compute
      tdm_load_2d(&Klds[buf ^ 1][0][0], Kh + (size_t)(k0 + 32) * HD_, 64, 32, HD_);
      tdm_load_2d(&Vlds[buf ^ 1][0][0], Vh + (k0 + 32), 32, 64, T_);
    }

    const bool active = (k0 <= qw + 15);
    if (active) {
      // S = Q Kt : two 16-col sets, two K(d)-chunks each (B-frags from LDS)
      v8f s0, s1;
#pragma unroll
      for (int e = 0; e < 8; ++e) { s0[e] = 0.0f; s1[e] = 0.0f; }
      {
        v16bf kb;
        kb = load_b_frag(&Klds[buf][0][0],   64, lr, hi);
        s0 = wmma_bf16(qf0, kb, s0);
        kb = load_b_frag(&Klds[buf][0][32],  64, lr, hi);
        s0 = wmma_bf16(qf1, kb, s0);
        kb = load_b_frag(&Klds[buf][16][0],  64, lr, hi);
        s1 = wmma_bf16(qf0, kb, s1);
        kb = load_b_frag(&Klds[buf][16][32], 64, lr, hi);
        s1 = wmma_bf16(qf1, kb, s1);
      }
      // causal mask on diagonal tiles
      if (k0 + 31 > qw) {
#pragma unroll
        for (int v = 0; v < 8; ++v) {
          const int q = qw + hi * 8 + v;
          if (k0 + lr > q)      s0[v] = -1e30f;
          if (k0 + 16 + lr > q) s1[v] = -1e30f;
        }
      }
      // online softmax, rows live across the 16-lane half-wave
      float alpha[8];
#pragma unroll
      for (int v = 0; v < 8; ++v) {
        float tm = fmaxf(s0[v], s1[v]);
        tm = fmaxf(tm, __shfl_xor(tm, 1, 32));
        tm = fmaxf(tm, __shfl_xor(tm, 2, 32));
        tm = fmaxf(tm, __shfl_xor(tm, 4, 32));
        tm = fmaxf(tm, __shfl_xor(tm, 8, 32));
        const float nm = fmaxf(run_m[v], tm);
        alpha[v] = __expf(run_m[v] - nm);
        run_m[v] = nm;
        const float p0 = __expf(s0[v] - nm);
        const float p1 = __expf(s1[v] - nm);
        float ts = p0 + p1;
        ts += __shfl_xor(ts, 1, 32);
        ts += __shfl_xor(ts, 2, 32);
        ts += __shfl_xor(ts, 4, 32);
        ts += __shfl_xor(ts, 8, 32);
        run_s[v] = run_s[v] * alpha[v] + ts;
        Plds[wave][hi * 8 + v][lr]      = f2bf(p0);
        Plds[wave][hi * 8 + v][16 + lr] = f2bf(p1);
      }
#pragma unroll
      for (int j = 0; j < 4; ++j)
#pragma unroll
        for (int e = 0; e < 8; ++e) oacc[j][e] *= alpha[e];
    }
    __syncthreads();  // order P store -> P load (block-uniform trip count)
    if (active) {
      V16U pu;
      pu.h[0] = *(const v8bf*)&Plds[wave][lr][hi * 8];
      pu.h[1] = *(const v8bf*)&Plds[wave][lr][16 + hi * 8];
      const v16bf pf = pu.v;
#pragma unroll
      for (int j = 0; j < 4; ++j) {
        v16bf vb = load_b_frag(&Vlds[buf][j * 16][0], 32, lr, hi);
        oacc[j] = wmma_bf16(pf, vb, oacc[j]);
      }
    }
    __syncthreads();  // protect Plds + K/V buffer before next iteration
  }

  const int b = bh / H_;
  const int h = bh % H_;
#pragma unroll
  for (int v = 0; v < 8; ++v) {
    const float inv = 1.0f / run_s[v];
    const int t = qw + hi * 8 + v;
    const size_t row = ((size_t)(b * T_ + t)) * C_ + h * HD_;
#pragma unroll
    for (int j = 0; j < 4; ++j)
      Obf[row + j * 16 + lr] = f2bf(oacc[j][v] * inv);
  }
}

// ---------------------------------------------------------------- launcher
extern "C" void kernel_launch(void* const* d_in, const int* in_sizes, int n_in,
                              void* d_out, int out_size, void* d_ws, size_t ws_size,
                              hipStream_t stream) {
  const float* x    = (const float*)d_in[0];
  const float* Wqkv = (const float*)d_in[1];
  const float* bqkv = (const float*)d_in[2];
  // d_in[3] = Wconv, d_in[4] = bconv: mathematically cancelled by softmax
  // shift-invariance (bias broadcasts over the key/softmax axis) -> skipped.
  const float* Wout = (const float*)d_in[5];
  const float* bout = (const float*)d_in[6];
  float* out = (float*)d_out;

  // workspace layout (bf16 = unsigned short), ~48 MB total
  unsigned short* xb  = (unsigned short*)d_ws;
  unsigned short* Wqb = xb  + (size_t)M_ * C_;          // 3072x1024
  unsigned short* Wob = Wqb + (size_t)NQKV * C_;        // 1024x1024
  unsigned short* Qb  = Wob + (size_t)C_ * C_;          // (B*H, T, 64)
  unsigned short* Kb  = Qb  + (size_t)B_ * H_ * T_ * HD_;
  unsigned short* Vt  = Kb  + (size_t)B_ * H_ * T_ * HD_;  // (B*H, 64, T)
  unsigned short* Obf = Vt  + (size_t)B_ * H_ * T_ * HD_;  // (B*T, C)

  k_f32_to_bf16<<<1024, 256, 0, stream>>>(x, xb, M_ * C_);
  k_f32_to_bf16<<<1024, 256, 0, stream>>>(Wqkv, Wqb, NQKV * C_);
  k_f32_to_bf16<<<512, 256, 0, stream>>>(Wout, Wob, C_ * C_);

  // QKV projection: (4096 x 1024) x (1024 x 3072)
  k_gemm<0><<<dim3(NQKV / 128, M_ / 128), 256, 0, stream>>>(
      xb, Wqb, bqkv, Qb, Kb, Vt, nullptr, C_, NQKV);

  // causal flash attention over 32 (b,h) heads
  k_attention<<<B_ * H_ * (T_ / 128), 256, 0, stream>>>(Qb, Kb, Vt, Obf);

  // output projection: (4096 x 1024) x (1024 x 1024) -> fp32 out
  k_gemm<1><<<dim3(C_ / 128, M_ / 128), 256, 0, stream>>>(
      Obf, Wob, bout, nullptr, nullptr, nullptr, out, C_, C_);

  (void)in_sizes; (void)n_in; (void)out_size; (void)ws_size;
}
